// TCGatherAttention_72413148611051
// MI455X (gfx1250) — matrix operand order repaired and checked by reading, not compiled, not run on hardware
//
#include <hip/hip_runtime.h>
#include <hip/hip_bf16.h>

// Problem constants
#define Bz    4
#define NQv   3136
#define NKVv  784
#define Cv    256
#define Hv    8
#define HDv   32
#define NGv   3136
#define SCALEv 0.17677669529663687f   // 1/sqrt(32)

typedef __attribute__((ext_vector_type(16))) __bf16          v16bf;
typedef __attribute__((ext_vector_type(16))) unsigned short  v16u;
typedef __attribute__((ext_vector_type(8)))  unsigned short  v8u;
typedef __attribute__((ext_vector_type(8)))  float           v8f;

// ---------- helpers ----------
__device__ __forceinline__ unsigned short f2bf(float f) {
    unsigned int u = __builtin_bit_cast(unsigned int, f);
    u += 0x7FFFu + ((u >> 16) & 1u);          // round-to-nearest-even
    return (unsigned short)(u >> 16);
}

__device__ __forceinline__ v16bf pack16(v8u lo, v8u hi) {
    v16u u;
#pragma unroll
    for (int i = 0; i < 8; ++i) { u[i] = lo[i]; u[8 + i] = hi[i]; }
    return __builtin_bit_cast(v16bf, u);
}

// A-fragment (16x32 bf16, MxK): lane holds row, 16 contiguous halves.
__device__ __forceinline__ v16bf load_a(const unsigned short* p) {
    v8u lo = *(const v8u*)p;
    v8u hi = *(const v8u*)(p + 8);
    return pack16(lo, hi);
}
// B-fragment (32x16 bf16, KxN) from K-contiguous storage: 16 contiguous halves.
__device__ __forceinline__ v16bf load_b(const unsigned short* p) {
    v8u lo = *(const v8u*)p;
    v8u hi = *(const v8u*)(p + 8);
    return pack16(lo, hi);
}

__device__ __forceinline__ v8f wmma_bf(v16bf a, v16bf b, v8f c) {
    return __builtin_amdgcn_wmma_f32_16x16x32_bf16(false, a, false, b, (short)0, c, false, false);
}

// ---------- elementwise / setup kernels ----------
__global__ void k_zero(float* p, int n) {
    int i = blockIdx.x * blockDim.x + threadIdx.x;
    if (i < n) p[i] = 0.f;
}

__global__ void k_f2bf(const float* in, unsigned short* out, int n) {
    int i = blockIdx.x * blockDim.x + threadIdx.x;
    if (i < n) out[i] = f2bf(in[i]);
}

// W (KxN, f32) -> WT (NxK, bf16)
__global__ void k_wtrans(const float* in, unsigned short* out, int K, int N) {
    int i = blockIdx.x * blockDim.x + threadIdx.x;
    if (i >= N * K) return;
    int n = i / K, k = i % K;
    out[i] = f2bf(in[k * N + n]);
}

// ---------- token_downup: scatter + normalize ----------
__global__ void k_scatter(const float* __restrict__ kv_in, const float* __restrict__ score,
                          const float* __restrict__ aggw, const int* __restrict__ idx_t,
                          const int* __restrict__ idx_s, float* __restrict__ acc) {
    int bp = blockIdx.x;                    // b*NG + p
    int b = bp / NGv;
    int t = idx_t[bp], s = idx_s[bp];
    float w = aggw[bp];
    int tid = threadIdx.x;
    float* dst = acc + (size_t)(b * NQv + t) * 258;
    float v = kv_in[(size_t)(b * NKVv + s) * Cv + tid] * w;
    atomicAdd(dst + tid, v);
    if (tid == 0) {
        atomicAdd(dst + 256, score[b * NKVv + s] * w);
        atomicAdd(dst + 257, w);
    }
}

__global__ void k_normalize(const float* __restrict__ acc, unsigned short* __restrict__ tmpkv,
                            float* __restrict__ conf) {
    int row = blockIdx.x, tid = threadIdx.x;
    const float* a = acc + (size_t)row * 258;
    float inv = 1.f / fmaxf(a[257], 1e-6f);
    tmpkv[(size_t)row * Cv + tid] = f2bf(a[tid] * inv);
    if (tid == 0) conf[row] = a[256] * inv;
}

// ---------- GEMM: q = q_in @ Wq  (bf16 out, no bias) ----------
__global__ void k_gemm_q(const unsigned short* __restrict__ A, const unsigned short* __restrict__ W,
                         unsigned short* __restrict__ Out) {
    int wid = blockIdx.x * 8 + (threadIdx.x >> 5);
    int lane = threadIdx.x & 31;
    int mt = wid >> 4, nt = wid & 15;
    const unsigned short* ap = A + (size_t)(mt * 16 + (lane & 15)) * 256 + ((lane >= 16) ? 16 : 0);
    const unsigned short* bp = W + (size_t)(nt * 16 + (lane & 15)) * 256 + ((lane >= 16) ? 16 : 0);
    v8f acc = {};
#pragma unroll
    for (int kc = 0; kc < 256; kc += 32)
        acc = wmma_bf(load_a(ap + kc), load_b(bp + kc), acc);
    int rbase = mt * 16 + ((lane >= 16) ? 8 : 0);
    int col = nt * 16 + (lane & 15);
#pragma unroll
    for (int r = 0; r < 8; ++r)
        Out[(size_t)(rbase + r) * 256 + col] = f2bf(acc[r]);
}

// ---------- GEMM + bias + LayerNorm + GELU (block = 16 rows x 256 cols) ----------
__global__ void k_gemm_g(const unsigned short* __restrict__ A, const unsigned short* __restrict__ W,
                         const float* __restrict__ bg, const float* __restrict__ lnw,
                         const float* __restrict__ lnb, unsigned short* __restrict__ Out) {
    __shared__ float smem[16 * 257];
    int tid = threadIdx.x, lane = tid & 31, w = tid >> 5;
    int m0 = blockIdx.x * 16;
    const unsigned short* ap = A + (size_t)(m0 + (lane & 15)) * 256 + ((lane >= 16) ? 16 : 0);
    int n0 = w * 32, n1 = w * 32 + 16;
    const unsigned short* b0p = W + (size_t)(n0 + (lane & 15)) * 256 + ((lane >= 16) ? 16 : 0);
    const unsigned short* b1p = W + (size_t)(n1 + (lane & 15)) * 256 + ((lane >= 16) ? 16 : 0);
    v8f a0 = {}, a1 = {};
#pragma unroll
    for (int kc = 0; kc < 256; kc += 32) {
        v16bf af = load_a(ap + kc);
        a0 = wmma_bf(af, load_b(b0p + kc), a0);
        a1 = wmma_bf(af, load_b(b1p + kc), a1);
    }
    int rbase = (lane >= 16) ? 8 : 0;
    int c0 = n0 + (lane & 15), c1 = n1 + (lane & 15);
#pragma unroll
    for (int r = 0; r < 8; ++r) {
        smem[(rbase + r) * 257 + c0] = a0[r] + bg[c0];
        smem[(rbase + r) * 257 + c1] = a1[r] + bg[c1];
    }
    __syncthreads();
    // LayerNorm + GELU: 16 threads per row (lanes 0-15 / 16-31 are distinct rows)
    int rr = tid >> 4, cc = tid & 15;
    float s = 0.f, s2 = 0.f;
#pragma unroll
    for (int j = 0; j < 16; ++j) {
        float x = smem[rr * 257 + cc + 16 * j];
        s += x; s2 += x * x;
    }
#pragma unroll
    for (int mm = 8; mm >= 1; mm >>= 1) { s += __shfl_xor(s, mm, 32); s2 += __shfl_xor(s2, mm, 32); }
    float mu = s * (1.f / 256.f);
    float var = fmaxf(s2 * (1.f / 256.f) - mu * mu, 0.f);
    float rstd = rsqrtf(var + 1e-5f);
#pragma unroll
    for (int j = 0; j < 16; ++j) {
        int col = cc + 16 * j;
        float x = smem[rr * 257 + col];
        float y = (x - mu) * rstd * lnw[col] + lnb[col];
        float g = 0.5f * y * (1.f + erff(y * 0.70710678118654752f));
        Out[(size_t)(m0 + rr) * 256 + col] = f2bf(g);
    }
}

// ---------- GEMM kv = kln @ Wkv (N=512) -> K (natural) and V (transposed (B,256,NQ)) ----------
__global__ void k_gemm_kv(const unsigned short* __restrict__ A, const unsigned short* __restrict__ W,
                          unsigned short* __restrict__ Kout, unsigned short* __restrict__ VT) {
    int wid = blockIdx.x * 8 + (threadIdx.x >> 5);
    int lane = threadIdx.x & 31;
    int mt = wid / 32, nt = wid % 32;
    const unsigned short* ap = A + (size_t)(mt * 16 + (lane & 15)) * 256 + ((lane >= 16) ? 16 : 0);
    const unsigned short* bp = W + (size_t)(nt * 16 + (lane & 15)) * 256 + ((lane >= 16) ? 16 : 0);
    v8f acc = {};
#pragma unroll
    for (int kc = 0; kc < 256; kc += 32)
        acc = wmma_bf(load_a(ap + kc), load_b(bp + kc), acc);
    int n = nt * 16 + (lane & 15);
    int rbase = mt * 16 + ((lane >= 16) ? 8 : 0);
#pragma unroll
    for (int r = 0; r < 8; ++r) {
        int row = rbase + r;
        unsigned short v = f2bf(acc[r]);
        if (n < 256) {
            Kout[(size_t)row * 256 + n] = v;
        } else {
            int b = row / NQv, q = row % NQv;
            VT[((size_t)b * 256 + (n - 256)) * NQv + q] = v;
        }
    }
}

// ---------- flash attention: one wave per (b,h,16-row q tile), stream 3136 keys ----------
__global__ void k_attn(const unsigned short* __restrict__ Q, const unsigned short* __restrict__ K,
                       const unsigned short* __restrict__ VT, const float* __restrict__ conf,
                       unsigned short* __restrict__ X) {
    __shared__ __align__(16) unsigned short pbuf[8][16 * 32];
    int tid = threadIdx.x, lane = tid & 31, w = tid >> 5;
    int g = blockIdx.x * 8 + w;
    int b = g / (Hv * (NQv / 16));
    int rem = g % (Hv * (NQv / 16));
    int h = rem / (NQv / 16);
    int qt = rem % (NQv / 16);

    int col = lane & 15;
    int kdb = (lane >= 16) ? 16 : 0;
    v16bf qf = load_a(Q + (size_t)(b * NQv + qt * 16 + col) * 256 + h * 32 + ((lane >= 16) ? 16 : 0));

    float m[8], l[8];
#pragma unroll
    for (int r = 0; r < 8; ++r) { m[r] = -1e30f; l[r] = 0.f; }
    v8f o0 = {}, o1 = {};
    v8f zf = {};
    const float* cf = conf + (size_t)b * NQv;

    for (int kc = 0; kc < NQv; kc += 32) {
        v16bf k0 = load_b(K + (size_t)(b * NQv + kc + col) * 256 + h * 32 + kdb);
        v16bf k1 = load_b(K + (size_t)(b * NQv + kc + 16 + col) * 256 + h * 32 + kdb);
        v8f s0 = wmma_bf(qf, k0, zf);        // Q(16x32) . K^T(32x16), K dim = HD = 32
        v8f s1 = wmma_bf(qf, k1, zf);
        float c0 = cf[kc + col], c1 = cf[kc + 16 + col];
        float p0[8], p1[8];
#pragma unroll
        for (int r = 0; r < 8; ++r) {
            float t0 = s0[r] * SCALEv + c0;
            float t1 = s1[r] * SCALEv + c1;
            float rm = fmaxf(t0, t1);
#pragma unroll
            for (int mm = 8; mm >= 1; mm >>= 1) rm = fmaxf(rm, __shfl_xor(rm, mm, 32));
            float mn = fmaxf(m[r], rm);
            float alpha = __expf(m[r] - mn);
            p0[r] = __expf(t0 - mn);
            p1[r] = __expf(t1 - mn);
            float ps = p0[r] + p1[r];
#pragma unroll
            for (int mm = 8; mm >= 1; mm >>= 1) ps += __shfl_xor(ps, mm, 32);
            l[r] = l[r] * alpha + ps;
            m[r] = mn;
            o0[r] *= alpha;
            o1[r] *= alpha;
        }
        // C-layout P -> row-major LDS -> A-fragment layout
        __syncthreads();                    // WAR guard vs previous iteration reads
        int rb = (lane >= 16) ? 8 : 0;
#pragma unroll
        for (int r = 0; r < 8; ++r) {
            pbuf[w][(rb + r) * 32 + col]      = f2bf(p0[r]);
            pbuf[w][(rb + r) * 32 + 16 + col] = f2bf(p1[r]);
        }
        __syncthreads();
        v16bf pf = load_a(&pbuf[w][col * 32 + ((lane >= 16) ? 16 : 0)]);
        v16bf v0 = load_b(VT + ((size_t)b * 256 + h * 32 + col) * NQv + kc + kdb);
        v16bf v1 = load_b(VT + ((size_t)b * 256 + h * 32 + 16 + col) * NQv + kc + kdb);
        o0 = wmma_bf(pf, v0, o0);           // P(16x32) . V(32x16) per HD half
        o1 = wmma_bf(pf, v1, o1);
    }
    int rbase = qt * 16 + ((lane >= 16) ? 8 : 0);
#pragma unroll
    for (int r = 0; r < 8; ++r) {
        float inv = 1.f / l[r];
        size_t row = (size_t)(b * NQv + rbase + r);
        X[row * 256 + h * 32 + col]      = f2bf(o0[r] * inv);
        X[row * 256 + h * 32 + 16 + col] = f2bf(o1[r] * inv);
    }
}

// ---------- final GEMM: out = x @ Wp + bp (f32 out) ----------
__global__ void k_gemm_out(const unsigned short* __restrict__ A, const unsigned short* __restrict__ W,
                           const float* __restrict__ bias, float* __restrict__ Out) {
    int wid = blockIdx.x * 8 + (threadIdx.x >> 5);
    int lane = threadIdx.x & 31;
    int mt = wid >> 4, nt = wid & 15;
    const unsigned short* ap = A + (size_t)(mt * 16 + (lane & 15)) * 256 + ((lane >= 16) ? 16 : 0);
    const unsigned short* bp = W + (size_t)(nt * 16 + (lane & 15)) * 256 + ((lane >= 16) ? 16 : 0);
    v8f acc = {};
#pragma unroll
    for (int kc = 0; kc < 256; kc += 32)
        acc = wmma_bf(load_a(ap + kc), load_b(bp + kc), acc);
    int rbase = mt * 16 + ((lane >= 16) ? 8 : 0);
    int cc = nt * 16 + (lane & 15);
    float bb = bias[cc];
#pragma unroll
    for (int r = 0; r < 8; ++r)
        Out[(size_t)(rbase + r) * 256 + cc] = acc[r] + bb;
}

extern "C" void kernel_launch(void* const* d_in, const int* in_sizes, int n_in,
                              void* d_out, int out_size, void* d_ws, size_t ws_size,
                              hipStream_t stream) {
    const float* q_in        = (const float*)d_in[0];
    const float* kv_in       = (const float*)d_in[1];
    const float* token_score = (const float*)d_in[2];
    const float* agg_weight  = (const float*)d_in[3];
    const int*   idx_t       = (const int*)d_in[4];
    const int*   idx_s       = (const int*)d_in[5];
    const float* Wq          = (const float*)d_in[6];
    const float* Wkv         = (const float*)d_in[7];
    const float* Wg          = (const float*)d_in[8];
    const float* bg          = (const float*)d_in[9];
    const float* ln_w        = (const float*)d_in[10];
    const float* ln_b        = (const float*)d_in[11];
    const float* Wp          = (const float*)d_in[12];
    const float* bp          = (const float*)d_in[13];
    float* out = (float*)d_out;

    char* ws = (char*)d_ws;
    size_t off = 0;
    auto alloc = [&](size_t bytes) -> void* {
        void* p = ws + off;
        off = (off + bytes + 255) & ~(size_t)255;
        return p;
    };
    const size_t M = (size_t)Bz * NQv;                    // 12544
    float*          acc      = (float*)alloc(M * 258 * 4);
    unsigned short* qin_bf   = (unsigned short*)alloc(M * 256 * 2);
    unsigned short* tmpkv_bf = (unsigned short*)alloc(M * 256 * 2);
    float*          conf     = (float*)alloc(M * 4);
    unsigned short* WqT      = (unsigned short*)alloc(256 * 256 * 2);
    unsigned short* WgT      = (unsigned short*)alloc(256 * 256 * 2);
    unsigned short* WkvT     = (unsigned short*)alloc(512 * 256 * 2);
    unsigned short* WpT      = (unsigned short*)alloc(256 * 256 * 2);
    unsigned short* q_bf     = (unsigned short*)alloc(M * 256 * 2);
    unsigned short* kln_bf   = (unsigned short*)alloc(M * 256 * 2);
    unsigned short* k_bf     = (unsigned short*)alloc(M * 256 * 2);
    unsigned short* vT_bf    = (unsigned short*)alloc((size_t)Bz * 256 * NQv * 2);
    unsigned short* x_bf     = (unsigned short*)alloc(M * 256 * 2);
    (void)ws_size; (void)in_sizes; (void)n_in; (void)out_size;   // ~59 MB used

    int nacc = Bz * NQv * 258;
    k_zero<<<(nacc + 255) / 256, 256, 0, stream>>>(acc, nacc);
    int nqin = (int)(M * 256);
    k_f2bf<<<(nqin + 255) / 256, 256, 0, stream>>>(q_in, qin_bf, nqin);
    k_wtrans<<<(256 * 256 + 255) / 256, 256, 0, stream>>>(Wq,  WqT,  256, 256);
    k_wtrans<<<(256 * 256 + 255) / 256, 256, 0, stream>>>(Wg,  WgT,  256, 256);
    k_wtrans<<<(512 * 256 + 255) / 256, 256, 0, stream>>>(Wkv, WkvT, 256, 512);
    k_wtrans<<<(256 * 256 + 255) / 256, 256, 0, stream>>>(Wp,  WpT,  256, 256);

    k_scatter<<<Bz * NGv, 256, 0, stream>>>(kv_in, token_score, agg_weight, idx_t, idx_s, acc);
    k_normalize<<<Bz * NQv, 256, 0, stream>>>(acc, tmpkv_bf, conf);

    k_gemm_q  <<<(int)(M / 16 * 16 / 8), 256, 0, stream>>>(qin_bf, WqT, q_bf);        // 12544 tiles
    k_gemm_g  <<<(int)(M / 16),          256, 0, stream>>>(tmpkv_bf, WgT, bg, ln_w, ln_b, kln_bf);
    k_gemm_kv <<<(int)(M / 16 * 32 / 8), 256, 0, stream>>>(kln_bf, WkvT, k_bf, vT_bf); // 25088 tiles
    k_attn    <<<Bz * Hv * (NQv / 16) / 8, 256, 0, stream>>>(q_bf, k_bf, vT_bf, conf, x_bf);
    k_gemm_out<<<(int)(M / 16 * 16 / 8), 256, 0, stream>>>(x_bf, WpT, bp, out);
}
